// MultiHeadedAttention_82051055223012
// MI455X (gfx1250) — matrix operand order repaired and checked by reading, not compiled, not run on hardware
//
#include <hip/hip_runtime.h>
#include <hip/hip_bf16.h>

// ---------------------------------------------------------------------------
// Relation-aware multi-head attention for MI455X (gfx1250), bf16 WMMA path.
// B=8, L=1024, D=512, H=8, d_k=64.
// ---------------------------------------------------------------------------

typedef __bf16  v16bf __attribute__((ext_vector_type(16)));
typedef float   v8f   __attribute__((ext_vector_type(8)));

#define B_  8
#define H_  8
#define L_  1024
#define D_  512
#define DK  64

#define SSTRIDE 1032                    // 1024 + 8 pad (breaks LDS bank conflicts)
#define SMEM_SS   (16 * SSTRIDE * 4)    // 66048 B fp32 scores/probs
#define SMEM_ACC  (16 * 64 * 4)         // 4096 B  fp32 PV accumulator
#define SMEM_RED  (16 * 8 * 4)          // 512 B   reduction partials
#define SMEM_MS   (32 * 4)              // sMax[16] + sSum[16]
#define SMEM_TOTAL (SMEM_SS + SMEM_ACC + SMEM_RED + SMEM_MS)

__device__ __forceinline__ unsigned short f2bf(float f) {
    unsigned int u = __builtin_bit_cast(unsigned int, f);
    u = (u + 0x7FFFu + ((u >> 16) & 1u)) >> 16;   // round-to-nearest-even
    return (unsigned short)u;
}

union Frag16 { v16bf v; uint4 u[2]; unsigned short s[16]; };

__device__ __forceinline__ v8f wmma_bf16(v16bf a, v16bf b, v8f c) {
    return __builtin_amdgcn_wmma_f32_16x16x32_bf16(
        /*neg_a=*/false, a, /*neg_b=*/false, b,
        /*c_mod=*/(short)0, c, /*reuse_a=*/false, /*reuse_b=*/false);
}

// ---------------------------------------------------------------------------
// fp32 -> bf16 conversion
// ---------------------------------------------------------------------------
__global__ __launch_bounds__(256) void cvt_f32_bf16(
    const float* __restrict__ src, unsigned short* __restrict__ dst, int n) {
    int i = blockIdx.x * 256 + threadIdx.x;
    if (i < n) dst[i] = f2bf(src[i]);
}

// ---------------------------------------------------------------------------
// out[M,N] = A[M,Dk] @ W[N,Dk]^T + bias[N]
//   M = 8192, N = D_ = 512, K = D_ = 512, bf16 operands, fp32 accumulate.
// Grid: (M/64, N/64), block 128 (4 waves). Wave w handles rows m0+16w..+15,
// all 64 columns of the workgroup's N tile.
// mode 0: write bf16 [B,H,L,64]      (Q / K)
// mode 1: write bf16 [B,H,64,L]      (V transposed)
// mode 2: write fp32 [M,N] plain     (final output)
// ---------------------------------------------------------------------------
__global__ __launch_bounds__(128) void gemm_bf16(
    const unsigned short* __restrict__ A,
    const unsigned short* __restrict__ W,
    const float* __restrict__ bias,
    void* __restrict__ out, int mode) {

    const int wave = threadIdx.x >> 5;
    const int lane = threadIdx.x & 31;
    const int m0 = blockIdx.x * 64 + wave * 16;
    const int n0 = blockIdx.y * 64;

    const int arow   = m0 + (lane & 15);
    const int achunk = (lane >> 4) * 8;      // ISA A layout: hi-lanes take K+8
    const unsigned short* ap = A + arow * D_;

    v8f acc[4];
    acc[0] = (v8f)0.f; acc[1] = (v8f)0.f; acc[2] = (v8f)0.f; acc[3] = (v8f)0.f;

    for (int k0 = 0; k0 < D_; k0 += 32) {
        Frag16 af;
        af.u[0] = *(const uint4*)(ap + k0 + achunk);        // K {0..7}/{8..15}
        af.u[1] = *(const uint4*)(ap + k0 + 16 + achunk);   // K {16..23}/{24..31}
#pragma unroll
        for (int nt = 0; nt < 4; ++nt) {
            const int wrow = n0 + nt * 16 + (lane & 15);
            const unsigned short* wp = W + wrow * D_ + k0 + (lane >> 4) * 8;
            Frag16 bf;
            bf.u[0] = *(const uint4*)(wp);
            bf.u[1] = *(const uint4*)(wp + 16);
            acc[nt] = wmma_bf16(af.v, bf.v, acc[nt]);
        }
    }

    // C layout: (VGPR r, lane) -> M = r + 8*(lane>=16), N = lane&15
    const int mbase = m0 + ((lane >> 4) << 3);
    const int nlane = lane & 15;
#pragma unroll
    for (int nt = 0; nt < 4; ++nt) {
        const int col = n0 + nt * 16 + nlane;
        const float bv = bias[col];
#pragma unroll
        for (int r = 0; r < 8; ++r) {
            const int row = mbase + r;
            const float val = acc[nt][r] + bv;
            if (mode == 2) {
                ((float*)out)[row * D_ + col] = val;
            } else {
                const int b = row >> 10, l = row & 1023;
                const int h = col >> 6,  d = col & 63;
                unsigned short* o = (unsigned short*)out;
                if (mode == 0)
                    o[((b * H_ + h) * L_ + l) * DK + d] = f2bf(val);   // [B,H,L,dk]
                else
                    o[((b * H_ + h) * DK + d) * L_ + l] = f2bf(val);   // [B,H,dk,L]
            }
        }
    }
}

// ---------------------------------------------------------------------------
// Attention core: one workgroup per (b, h, 16-row query tile).
// Grid ordering: h fastest -> adjacent WGs share tree/leaf/mask rows in L2.
// ---------------------------------------------------------------------------
__global__ __launch_bounds__(128) void attn_kernel(
    const unsigned short* __restrict__ Q,     // [B,H,L,64]  bf16
    const unsigned short* __restrict__ K,     // [B,H,L,64]  bf16
    const unsigned short* __restrict__ Vt,    // [B,H,64,L]  bf16
    const float* __restrict__ tree,           // [B,L,L]
    const float* __restrict__ leaf,           // [B,L,L]
    const int*   __restrict__ mask,           // [B,1,L,L]
    unsigned short* __restrict__ X) {         // [B,L,512]   bf16

    extern __shared__ char smem[];
    float* sS   = (float*)smem;                               // [16][SSTRIDE]
    float* sAcc = (float*)(smem + SMEM_SS);                   // [16][64]
    float* sRed = (float*)(smem + SMEM_SS + SMEM_ACC);        // [16][8]
    float* sMax = (float*)(smem + SMEM_SS + SMEM_ACC + SMEM_RED);
    float* sSum = sMax + 16;

    const int wg = blockIdx.x;
    const int h  = wg & 7;
    const int b  = (wg >> 3) & 7;
    const int qt = wg >> 6;
    const int bh = wg & 63;                  // b*H + h
    const int q0 = qt * 16;

    const int tid  = threadIdx.x;
    const int wave = tid >> 5;
    const int lane = tid & 31;

    // ---- Q tile A-fragments (16x64 -> two 16x32 fragments) ----
    const int arow   = lane & 15;
    const int achunk = (lane >> 4) * 8;
    const unsigned short* qp = Q + (bh * L_ + q0 + arow) * DK;
    Frag16 a0, a1;
    a0.u[0] = *(const uint4*)(qp + achunk);
    a0.u[1] = *(const uint4*)(qp + 16 + achunk);
    a1.u[0] = *(const uint4*)(qp + 32 + achunk);
    a1.u[1] = *(const uint4*)(qp + 48 + achunk);

    // ---- scores: each wave owns 16 key tiles (256 keys) ----
    const unsigned short* kb = K + bh * L_ * DK;
    for (int t = 0; t < 16; ++t) {
        const int k0 = (wave * 16 + t) * 16;
        const unsigned short* kp = kb + (k0 + (lane & 15)) * DK + (lane >> 4) * 8;
        Frag16 b0, b1;
        b0.u[0] = *(const uint4*)(kp);        b0.u[1] = *(const uint4*)(kp + 16);
        b1.u[0] = *(const uint4*)(kp + 32);   b1.u[1] = *(const uint4*)(kp + 48);
        v8f c = (v8f)0.f;
        c = wmma_bf16(a0.v, b0.v, c);
        c = wmma_bf16(a1.v, b1.v, c);
        const int m8 = (lane >> 4) * 8, n = lane & 15;
#pragma unroll
        for (int r = 0; r < 8; ++r) sS[(m8 + r) * SSTRIDE + k0 + n] = c[r];
    }
    __syncthreads();

    // ---- scale + tree/leaf bias + mask (coalesced) ----
    const float scale = 0.125f;                     // 1/sqrt(64)
    const float* trow = tree + (size_t)(b * L_ + q0) * L_;
    const float* lrow = leaf + (size_t)(b * L_ + q0) * L_;
    const int*   mrow = mask + (size_t)(b * L_ + q0) * L_;
    for (int idx = tid; idx < 16 * L_; idx += 128) {
        const int row = idx >> 10, col = idx & 1023;
        float v = sS[row * SSTRIDE + col] * scale + trow[row * L_ + col] + lrow[row * L_ + col];
        if (mrow[row * L_ + col] == 0) v = -1e9f;
        sS[row * SSTRIDE + col] = v;
    }
    __syncthreads();

    // ---- row max (8 threads per row) ----
    {
        const int row = tid >> 3, j = tid & 7;
        float m = -3.4e38f;
        for (int c = j; c < L_; c += 8) m = fmaxf(m, sS[row * SSTRIDE + c]);
        sRed[row * 8 + j] = m;
    }
    __syncthreads();
    if (tid < 16) {
        float m = sRed[tid * 8];
#pragma unroll
        for (int j = 1; j < 8; ++j) m = fmaxf(m, sRed[tid * 8 + j]);
        sMax[tid] = m;
    }
    __syncthreads();

    // ---- exp + row sum ----
    {
        const int row = tid >> 3, j = tid & 7;
        const float mx = sMax[row];
        float s = 0.f;
        for (int c = j; c < L_; c += 8) {
            const float e = __expf(sS[row * SSTRIDE + c] - mx);
            sS[row * SSTRIDE + c] = e;
            s += e;
        }
        sRed[row * 8 + j] = s;
    }
    __syncthreads();
    if (tid < 16) {
        float s = 0.f;
#pragma unroll
        for (int j = 0; j < 8; ++j) s += sRed[tid * 8 + j];
        sSum[tid] = 1.0f / s;
    }
    for (int i = tid; i < 16 * 64; i += 128) sAcc[i] = 0.f;
    __syncthreads();

    // ---- PV: wave consumes its 256 keys in 8 steps of K=32 ----
    const unsigned short* vb = Vt + bh * DK * L_;
    const int prow   = lane & 15;
    const int pchunk = (lane >> 4) * 8;
    const float rinv = sSum[prow];
    v8f acc[4];
    acc[0] = (v8f)0.f; acc[1] = (v8f)0.f; acc[2] = (v8f)0.f; acc[3] = (v8f)0.f;

    for (int kk = 0; kk < 8; ++kk) {
        const int kbase = wave * 256 + kk * 32;
        Frag16 af;
#pragma unroll
        for (int i = 0; i < 8; ++i)
            af.s[i] = f2bf(sS[prow * SSTRIDE + kbase + pchunk + i] * rinv);
#pragma unroll
        for (int i = 0; i < 8; ++i)
            af.s[8 + i] = f2bf(sS[prow * SSTRIDE + kbase + 16 + pchunk + i] * rinv);
#pragma unroll
        for (int nt = 0; nt < 4; ++nt) {
            const unsigned short* vp = vb + (nt * 16 + (lane & 15)) * L_
                                          + kbase + (lane >> 4) * 8;
            Frag16 bf;
            bf.u[0] = *(const uint4*)(vp);
            bf.u[1] = *(const uint4*)(vp + 16);
            acc[nt] = wmma_bf16(af.v, bf.v, acc[nt]);
        }
    }

    // ---- cross-wave combine via LDS float atomics ----
    {
        const int m8 = (lane >> 4) * 8, n = lane & 15;
#pragma unroll
        for (int nt = 0; nt < 4; ++nt)
#pragma unroll
            for (int r = 0; r < 8; ++r)
                atomicAdd(&sAcc[(m8 + r) * 64 + nt * 16 + n], acc[nt][r]);
    }
    __syncthreads();

    // ---- write X tile: [B,L,512] bf16, head offset h*64 ----
    unsigned short* xp = X + (size_t)(b * L_ + q0) * D_ + h * DK;
    for (int i = tid; i < 16 * 64; i += 128) {
        const int row = i >> 6, col = i & 63;
        xp[row * D_ + col] = f2bf(sAcc[i]);
    }
}

// ---------------------------------------------------------------------------
// Host launch
// ---------------------------------------------------------------------------
extern "C" void kernel_launch(void* const* d_in, const int* in_sizes, int n_in,
                              void* d_out, int out_size, void* d_ws, size_t ws_size,
                              hipStream_t stream) {
    (void)in_sizes; (void)n_in; (void)out_size; (void)ws_size;

    const float* query = (const float*)d_in[0];
    const float* key_  = (const float*)d_in[1];
    const float* value = (const float*)d_in[2];
    const int*   mask  = (const int*)  d_in[3];
    const float* tree  = (const float*)d_in[4];
    const float* leaf  = (const float*)d_in[5];
    const float* Wq = (const float*)d_in[6];  const float* bq = (const float*)d_in[7];
    const float* Wk = (const float*)d_in[8];  const float* bk = (const float*)d_in[9];
    const float* Wv = (const float*)d_in[10]; const float* bv = (const float*)d_in[11];
    const float* Wo = (const float*)d_in[12]; const float* bo = (const float*)d_in[13];

    const size_t BLD = (size_t)B_ * L_ * D_;   // 4194304
    const size_t DD  = (size_t)D_ * D_;        // 262144

    // workspace carve (bf16 = 2 bytes, all chunks 256B aligned by construction)
    unsigned short* p = (unsigned short*)d_ws;
    unsigned short* qin = p;            p += BLD;
    unsigned short* kin = p;            p += BLD;
    unsigned short* vin = p;            p += BLD;
    unsigned short* wqb = p;            p += DD;
    unsigned short* wkb = p;            p += DD;
    unsigned short* wvb = p;            p += DD;
    unsigned short* wob = p;            p += DD;
    unsigned short* Qb  = p;            p += BLD;
    unsigned short* Kb  = p;            p += BLD;
    unsigned short* Vtb = p;            p += BLD;
    unsigned short* Xb  = p;            p += BLD;

    // 1) fp32 -> bf16
    cvt_f32_bf16<<<(int)((BLD + 255) / 256), 256, 0, stream>>>(query, qin, (int)BLD);
    cvt_f32_bf16<<<(int)((BLD + 255) / 256), 256, 0, stream>>>(key_,  kin, (int)BLD);
    cvt_f32_bf16<<<(int)((BLD + 255) / 256), 256, 0, stream>>>(value, vin, (int)BLD);
    cvt_f32_bf16<<<(int)((DD  + 255) / 256), 256, 0, stream>>>(Wq, wqb, (int)DD);
    cvt_f32_bf16<<<(int)((DD  + 255) / 256), 256, 0, stream>>>(Wk, wkb, (int)DD);
    cvt_f32_bf16<<<(int)((DD  + 255) / 256), 256, 0, stream>>>(Wv, wvb, (int)DD);
    cvt_f32_bf16<<<(int)((DD  + 255) / 256), 256, 0, stream>>>(Wo, wob, (int)DD);

    // 2) projections (M=8192, N=512): Q,K plain [B,H,L,64]; V transposed [B,H,64,L]
    dim3 gg(B_ * L_ / 64, D_ / 64);
    gemm_bf16<<<gg, 128, 0, stream>>>(qin, wqb, bq, (void*)Qb,  0);
    gemm_bf16<<<gg, 128, 0, stream>>>(kin, wkb, bk, (void*)Kb,  0);
    gemm_bf16<<<gg, 128, 0, stream>>>(vin, wvb, bv, (void*)Vtb, 1);

    // 3) attention: one WG per (b,h,16-query tile); h fastest for L2 bias reuse
    attn_kernel<<<B_ * H_ * (L_ / 16), 128, SMEM_TOTAL, stream>>>(
        Qb, Kb, Vtb, tree, leaf, mask, Xb);

    // 4) output projection -> fp32 d_out
    gemm_bf16<<<gg, 128, 0, stream>>>(Xb, wob, bo, d_out, 2);
}
